// APPNPM_15479062135294
// MI455X (gfx1250) — compile-verified, matrix operand order checked by reading, not run on hardware
//
#include <hip/hip_runtime.h>
#include <hip/hip_bf16.h>

// ---------------------------------------------------------------------------
// APPNP on MI455X (gfx1250): bf16 WMMA MLP (software-pipelined k-loop)
//                            + L2-resident CSR propagation
// ---------------------------------------------------------------------------

#define N_NODES   100000
#define N_EDGES   3200000
#define IN_FEATS  512
#define N_HIDDEN  256
#define N_CLASSES 64
#define K_STEPS   10
#define ALPHA_F   0.2f
#define ONE_M_ALPHA_F 0.8f

typedef __attribute__((ext_vector_type(16))) __bf16 v16bf;
typedef __attribute__((ext_vector_type(8)))  __bf16 v8bf;
typedef __attribute__((ext_vector_type(8)))  float  v8f;

// ---- fragment loaders ------------------------------------------------------
// 16-bit WMMA A/B layout (wave32): lane L holds row (A) / col (B)  r = L%16,
// K chunks [kb, kb+8) and [kb+16, kb+24) with kb = (L<16 ? 0 : 8).
// 'p' points at element (r, k + kb); both chunks are 16B-aligned.

__device__ __forceinline__ v16bf load_frag_bf16(const __bf16* p) {
    v8bf c0 = *(const v8bf*)(p);
    v8bf c1 = *(const v8bf*)(p + 16);
    return __builtin_shufflevector(c0, c1,
        0, 1, 2, 3, 4, 5, 6, 7, 8, 9, 10, 11, 12, 13, 14, 15);
}

__device__ __forceinline__ v16bf load_frag_f32(const float* p) {
    v16bf a;
#pragma unroll
    for (int i = 0; i < 8; ++i) a[i]     = (__bf16)p[i];
#pragma unroll
    for (int i = 0; i < 8; ++i) a[8 + i] = (__bf16)p[16 + i];
    return a;
}

// ---- GEMM: out[M x N] = act(A[M x K] @ Wt[N x K]^T + bias) -----------------
// One wave computes a 16x64 strip (4 WMMA tiles along N, A fragment reused).
// Software-pipelined: fragments for k+32 are issued before the WMMA chain of
// k so loads overlap the matrix pipe. Wave-uniform m-tile guard keeps EXEC
// all-ones inside WMMA (ISA requirement).

template <int K, int N, bool A_F32, bool RELU, bool OUT_BF16>
__global__ __launch_bounds__(256) void gemm_wmma_kernel(
    const void* __restrict__ A_, const __bf16* __restrict__ Wt,
    const float* __restrict__ bias, void* __restrict__ out, int Mtiles)
{
    const int wave   = threadIdx.x >> 5;
    const int lane   = threadIdx.x & 31;
    const int m_tile = blockIdx.x * 8 + wave;
    if (m_tile >= Mtiles) return;               // wave-uniform

    const int n0  = blockIdx.y * 64;
    const int r   = lane & 15;                  // row (A) / col (B) in tile
    const int hi8 = (lane >> 4) << 3;           // K-chunk base: 0 or 8
    const int m   = m_tile * 16 + r;

    const float*  Af = (const float*)A_;
    const __bf16* Ab = (const __bf16*)A_;

    v8f zero;
#pragma unroll
    for (int i = 0; i < 8; ++i) zero[i] = 0.0f;
    v8f acc[4] = { zero, zero, zero, zero };

    // prologue: fragments for k = 0
    v16bf a_cur;
    if constexpr (A_F32) a_cur = load_frag_f32 (Af + (size_t)m * K + hi8);
    else                 a_cur = load_frag_bf16(Ab + (size_t)m * K + hi8);
    v16bf b_cur[4];
#pragma unroll
    for (int nt = 0; nt < 4; ++nt)
        b_cur[nt] = load_frag_bf16(Wt + (size_t)(n0 + nt * 16 + r) * K + hi8);

#pragma unroll
    for (int k = 0; k < K - 32; k += 32) {
        // issue next k-step's loads before this step's WMMAs
        v16bf a_nxt;
        if constexpr (A_F32) a_nxt = load_frag_f32 (Af + (size_t)m * K + (k + 32) + hi8);
        else                 a_nxt = load_frag_bf16(Ab + (size_t)m * K + (k + 32) + hi8);
        v16bf b_nxt[4];
#pragma unroll
        for (int nt = 0; nt < 4; ++nt)
            b_nxt[nt] = load_frag_bf16(Wt + (size_t)(n0 + nt * 16 + r) * K + (k + 32) + hi8);

#pragma unroll
        for (int nt = 0; nt < 4; ++nt)
            acc[nt] = __builtin_amdgcn_wmma_f32_16x16x32_bf16(
                false, a_cur, false, b_cur[nt], (short)0, acc[nt], false, false);

        a_cur = a_nxt;
#pragma unroll
        for (int nt = 0; nt < 4; ++nt) b_cur[nt] = b_nxt[nt];
    }

    // epilogue: last k-step
#pragma unroll
    for (int nt = 0; nt < 4; ++nt)
        acc[nt] = __builtin_amdgcn_wmma_f32_16x16x32_bf16(
            false, a_cur, false, b_cur[nt], (short)0, acc[nt], false, false);

    // C/D layout: VGPR j -> (M = j + hi8, N = r)
#pragma unroll
    for (int nt = 0; nt < 4; ++nt) {
        const int   col = n0 + nt * 16 + r;
        const float bv  = bias[col];
#pragma unroll
        for (int j = 0; j < 8; ++j) {
            float v = acc[nt][j] + bv;
            if constexpr (RELU) v = fmaxf(v, 0.0f);
            const int orow = m_tile * 16 + hi8 + j;
            if constexpr (OUT_BF16)
                ((__bf16*)out)[(size_t)orow * N + col] = (__bf16)v;
            else
                ((float*)out)[(size_t)orow * N + col] = v;
        }
    }
}

// ---- weight transpose + bf16 convert: Wt[n][k] = (bf16) W[k][n] ------------
__global__ void transpose_bf16_kernel(const float* __restrict__ W,
                                      __bf16* __restrict__ Wt, int K, int N) {
    int idx = blockIdx.x * blockDim.x + threadIdx.x;
    if (idx >= K * N) return;
    int k = idx / N, n = idx % N;
    Wt[(size_t)n * K + k] = (__bf16)W[(size_t)k * N + n];
}

// ---- graph preprocessing ---------------------------------------------------
__global__ void count_kernel(const int* __restrict__ col, int* __restrict__ cnt) {
    int e = blockIdx.x * blockDim.x + threadIdx.x;
    if (e < N_EDGES) atomicAdd(&cnt[col[e]], 1);
}

__global__ void dinv_kernel(const int* __restrict__ cnt, float* __restrict__ dinv) {
    int i = blockIdx.x * blockDim.x + threadIdx.x;
    if (i < N_NODES) dinv[i] = rsqrtf((float)cnt[i] + 1.0f);  // +1 self loop
}

// block-local exclusive scan (Hillis-Steele) + block partial sums
__global__ void scan_local_kernel(const int* __restrict__ cnt,
                                  int* __restrict__ row_ptr,
                                  int* __restrict__ partials, int n) {
    __shared__ int sm[256];
    int i = blockIdx.x * 256 + threadIdx.x;
    int v = (i < n) ? cnt[i] : 0;
    sm[threadIdx.x] = v;
    __syncthreads();
#pragma unroll
    for (int off = 1; off < 256; off <<= 1) {
        int t = (threadIdx.x >= off) ? sm[threadIdx.x - off] : 0;
        __syncthreads();
        sm[threadIdx.x] += t;
        __syncthreads();
    }
    if (i < n) row_ptr[i] = sm[threadIdx.x] - v;   // exclusive
    if (threadIdx.x == 255) partials[blockIdx.x] = sm[255];
}

__global__ void scan_partials_kernel(int* __restrict__ partials, int nb) {
    if (blockIdx.x == 0 && threadIdx.x == 0) {
        int run = 0;
        for (int b = 0; b < nb; ++b) { int v = partials[b]; partials[b] = run; run += v; }
    }
}

__global__ void add_offsets_kernel(int* __restrict__ row_ptr,
                                   const int* __restrict__ partials,
                                   int* __restrict__ nxt) {
    int i = blockIdx.x * blockDim.x + threadIdx.x;
    if (i >= N_NODES) return;
    int v = row_ptr[i] + partials[i >> 8];
    row_ptr[i] = v;
    nxt[i] = v;
}

__global__ void fill_kernel(const int* __restrict__ row, const int* __restrict__ col,
                            const float* __restrict__ dinv, int* __restrict__ nxt,
                            int* __restrict__ src, float* __restrict__ wgt) {
    int e = blockIdx.x * blockDim.x + threadIdx.x;
    if (e >= N_EDGES) return;
    int c = col[e], r = row[e];
    int p = atomicAdd(&nxt[c], 1);
    src[p] = r;
    wgt[p] = dinv[r] * dinv[c];
}

__global__ void copy_kernel(const float* __restrict__ a, float* __restrict__ b, int n) {
    int i = blockIdx.x * blockDim.x + threadIdx.x;
    if (i < n) b[i] = a[i];
}

// ---- APPNP step: z' = (1-a)*(sum_in w*z[src] + dinv^2*z) + a*h -------------
// 16 threads per node, float4 channels each; edge metadata broadcast within
// the half-wave; z-row gathers are 256B coalesced. Working set ~130MB -> L2.
__global__ __launch_bounds__(256) void prop_kernel(
    const float* __restrict__ z, const int* __restrict__ row_ptr,
    const int* __restrict__ cnt, const int* __restrict__ src,
    const float* __restrict__ wgt, const float* __restrict__ dinv,
    const float* __restrict__ h, float* __restrict__ zout)
{
    int t = blockIdx.x * blockDim.x + threadIdx.x;
    if (t >= N_NODES * 16) return;
    int i  = t >> 4;
    int c0 = (t & 15) << 2;

    float4 acc = make_float4(0.f, 0.f, 0.f, 0.f);
    int s = row_ptr[i];
    int e = s + cnt[i];
    for (int j = s; j < e; ++j) {
        int   sn = src[j];
        float wv = wgt[j];
        const float4 zz = *(const float4*)(z + (size_t)sn * 64 + c0);
        acc.x += wv * zz.x; acc.y += wv * zz.y;
        acc.z += wv * zz.z; acc.w += wv * zz.w;
    }
    float di = dinv[i];
    float sw = di * di;
    const float4 zi = *(const float4*)(z + (size_t)i * 64 + c0);
    const float4 hh = *(const float4*)(h + (size_t)i * 64 + c0);
    float4 o;
    o.x = ONE_M_ALPHA_F * (acc.x + sw * zi.x) + ALPHA_F * hh.x;
    o.y = ONE_M_ALPHA_F * (acc.y + sw * zi.y) + ALPHA_F * hh.y;
    o.z = ONE_M_ALPHA_F * (acc.z + sw * zi.z) + ALPHA_F * hh.z;
    o.w = ONE_M_ALPHA_F * (acc.w + sw * zi.w) + ALPHA_F * hh.w;
    *(float4*)(zout + (size_t)i * 64 + c0) = o;
}

// ---------------------------------------------------------------------------
// Workspace layout (bytes, all offsets 16B-aligned; ~130 MB total)
// ---------------------------------------------------------------------------
#define OFF_H     ((size_t)0)                       // h f32      [N,64]  25.6MB
#define OFF_H1    ((size_t)25600000)                // h1 bf16    [N,256] 51.2MB
#define OFF_H2    ((size_t)76800000)                // h2 bf16    [N,256] 51.2MB
#define OFF_Z0    OFF_H1                            // reuse h1 region (dead after gemm2)
#define OFF_Z1    ((size_t)51200000)
#define OFF_SRC   OFF_H2                            // reuse h2 region (dead after gemm3)
#define OFF_WGT   ((size_t)89600000)
#define OFF_CNT   ((size_t)128000000)
#define OFF_RP    ((size_t)128400000)
#define OFF_NEXT  ((size_t)128800000)
#define OFF_DINV  ((size_t)129200000)
#define OFF_PART  ((size_t)129600000)
#define OFF_W0T   ((size_t)129604096)               // 512*256*2 = 262144
#define OFF_W1T   ((size_t)129866240)               // 256*256*2 = 131072
#define OFF_W2T   ((size_t)129997312)               // 256*64*2  =  32768

extern "C" void kernel_launch(void* const* d_in, const int* in_sizes, int n_in,
                              void* d_out, int out_size, void* d_ws, size_t ws_size,
                              hipStream_t stream) {
    (void)in_sizes; (void)n_in; (void)out_size; (void)ws_size;

    const float* x   = (const float*)d_in[0];
    const int*   ei  = (const int*)d_in[1];
    const int*   row = ei;                // edge_index[0] : sources
    const int*   col = ei + N_EDGES;      // edge_index[1] : targets
    const float* W0  = (const float*)d_in[2];
    const float* b0  = (const float*)d_in[3];
    const float* W1  = (const float*)d_in[4];
    const float* b1  = (const float*)d_in[5];
    const float* W2  = (const float*)d_in[6];
    const float* b2  = (const float*)d_in[7];
    float*       out = (float*)d_out;

    char* ws = (char*)d_ws;
    float*  h    = (float*)(ws + OFF_H);
    __bf16* h1   = (__bf16*)(ws + OFF_H1);
    __bf16* h2   = (__bf16*)(ws + OFF_H2);
    float*  z0   = (float*)(ws + OFF_Z0);
    float*  z1   = (float*)(ws + OFF_Z1);
    int*    srcA = (int*)(ws + OFF_SRC);
    float*  wgt  = (float*)(ws + OFF_WGT);
    int*    cnt  = (int*)(ws + OFF_CNT);
    int*    rp   = (int*)(ws + OFF_RP);
    int*    nxt  = (int*)(ws + OFF_NEXT);
    float*  dinv = (float*)(ws + OFF_DINV);
    int*    part = (int*)(ws + OFF_PART);
    __bf16* W0t  = (__bf16*)(ws + OFF_W0T);
    __bf16* W1t  = (__bf16*)(ws + OFF_W1T);
    __bf16* W2t  = (__bf16*)(ws + OFF_W2T);

    // --- weight prep (transpose + bf16) ---
    transpose_bf16_kernel<<<(IN_FEATS * N_HIDDEN + 255) / 256, 256, 0, stream>>>(W0, W0t, IN_FEATS, N_HIDDEN);
    transpose_bf16_kernel<<<(N_HIDDEN * N_HIDDEN + 255) / 256, 256, 0, stream>>>(W1, W1t, N_HIDDEN, N_HIDDEN);
    transpose_bf16_kernel<<<(N_HIDDEN * N_CLASSES + 255) / 256, 256, 0, stream>>>(W2, W2t, N_HIDDEN, N_CLASSES);

    // --- MLP via WMMA ---
    const int Mtiles = N_NODES / 16;                // 6250
    const int gx     = (Mtiles + 7) / 8;            // 782
    gemm_wmma_kernel<IN_FEATS, N_HIDDEN, true,  true,  true >
        <<<dim3(gx, N_HIDDEN / 64), 256, 0, stream>>>(x,  W0t, b0, h1, Mtiles);
    gemm_wmma_kernel<N_HIDDEN, N_HIDDEN, false, true,  true >
        <<<dim3(gx, N_HIDDEN / 64), 256, 0, stream>>>(h1, W1t, b1, h2, Mtiles);
    gemm_wmma_kernel<N_HIDDEN, N_CLASSES, false, false, false>
        <<<dim3(gx, 1),             256, 0, stream>>>(h2, W2t, b2, h,  Mtiles);

    // --- GCN norm + CSR build (by destination) ---
    hipMemsetAsync(cnt, 0, (size_t)N_NODES * sizeof(int), stream);
    count_kernel<<<(N_EDGES + 255) / 256, 256, 0, stream>>>(col, cnt);
    dinv_kernel<<<(N_NODES + 255) / 256, 256, 0, stream>>>(cnt, dinv);

    const int nScanBlocks = (N_NODES + 255) / 256;  // 391
    scan_local_kernel<<<nScanBlocks, 256, 0, stream>>>(cnt, rp, part, N_NODES);
    scan_partials_kernel<<<1, 1, 0, stream>>>(part, nScanBlocks);
    add_offsets_kernel<<<(N_NODES + 255) / 256, 256, 0, stream>>>(rp, part, nxt);
    fill_kernel<<<(N_EDGES + 255) / 256, 256, 0, stream>>>(row, col, dinv, nxt, srcA, wgt);

    // --- APPNP propagation (L2-resident working set) ---
    copy_kernel<<<(N_NODES * 64 + 255) / 256, 256, 0, stream>>>(h, z0, N_NODES * 64);
    const float* cur = z0;
    for (int k = 0; k < K_STEPS; ++k) {
        float* dst = (k == K_STEPS - 1) ? out : ((k & 1) ? z0 : z1);
        prop_kernel<<<(N_NODES * 16 + 255) / 256, 256, 0, stream>>>(
            cur, rp, cnt, srcA, wgt, dinv, h, dst);
        cur = dst;
    }
}
